// TemporalEncoding_65609920414079
// MI455X (gfx1250) — matrix operand order repaired and checked by reading, not compiled, not run on hardware
//
#include <hip/hip_runtime.h>
#include <hip/hip_bf16.h>

// Problem constants (match reference)
#define TT 64
#define NN 2048
#define EE 8192
#define DD 128
#define HH 12
#define DH 10
#define HD 120

typedef __attribute__((ext_vector_type(16))) _Float16 v16h;
typedef __attribute__((ext_vector_type(8)))  _Float16 v8h;
typedef __attribute__((ext_vector_type(4)))  _Float16 v4h;
typedef __attribute__((ext_vector_type(8)))  float    v8f;
typedef __attribute__((ext_vector_type(4)))  unsigned int u32x4;
typedef __attribute__((ext_vector_type(8)))  unsigned int u32x8;

// Use the inline-asm TDM path (portable across both toolchains per bridge doc).
#define USE_TDM_ASM 1

// ---------------------------------------------------------------------------
// WMMA helpers (CDNA5 v_wmma_f32_16x16x32_f16, wave32)
// ---------------------------------------------------------------------------
__device__ __forceinline__ v8f wmma_f16(v16h a, v16h b, v8f c) {
  // 8 args: (neg_a, A, neg_b, B, c_mod, C, reuse_a, reuse_b)
  return __builtin_amdgcn_wmma_f32_16x16x32_f16(false, a, false, b, (short)0, c,
                                                false, false);
}

// A fragment: 16x32 (MxK) f16, row-major source with leading dim `lda` halves.
// Lane l: row M = m0 + (l&15); K-half = l>>4.
//   elems 0..7  -> K = k0 + kh*16 .. +7
//   elems 8..15 -> K = k0 + kh*16 + 8 .. +7   (contiguous, matches B)
__device__ __forceinline__ v16h load_a_frag(const _Float16* A, int lane,
                                            int m0, int k0, int lda) {
  const int m  = m0 + (lane & 15);
  const int kh = lane >> 4;
  const _Float16* base = A + (size_t)m * lda + k0 + kh * 16;
  v8h lo = *(const v8h*)(base);
  v8h hi = *(const v8h*)(base + 8);
  v16h a;
#pragma unroll
  for (int j = 0; j < 8; ++j) { a[j] = lo[j]; a[j + 8] = hi[j]; }
  return a;
}

// B fragment: 32x16 (KxN) f16 from N-major storage Bt[n][k] (ldb halves).
// Lane l: col N = n0 + (l&15); K run = k0 + (l>>4)*16 .. +15 (contiguous).
__device__ __forceinline__ v16h load_bt_frag(const _Float16* Bt, int lane,
                                             int n0, int k0, int ldb) {
  const int n = n0 + (lane & 15);
  return *(const v16h*)(Bt + (size_t)n * ldb + k0 + (lane >> 4) * 16);
}

// pack a float4 into 4 halves and store as one 64-bit LDS write
__device__ __forceinline__ void store4h(_Float16* dst, float4 f) {
  v4h p;
  p[0] = (_Float16)f.x; p[1] = (_Float16)f.y;
  p[2] = (_Float16)f.z; p[3] = (_Float16)f.w;
  *(v4h*)dst = p;
}

// ---------------------------------------------------------------------------
// 0) Weight prep: f32 -> f16, transpose to N-major, pad 120->128, zero agg
// ---------------------------------------------------------------------------
__global__ void prep_kernel(const float* __restrict__ Wp, const float* __restrict__ bp,
                            const float* __restrict__ Wq, const float* __restrict__ bq,
                            const float* __restrict__ Wk, const float* __restrict__ bk,
                            const float* __restrict__ Wv, const float* __restrict__ bv,
                            _Float16* __restrict__ WpT, _Float16* __restrict__ WqT,
                            _Float16* __restrict__ WkT, _Float16* __restrict__ WvT,
                            float* __restrict__ bpP, float* __restrict__ bqP,
                            float* __restrict__ bkP, float* __restrict__ bvP,
                            float* __restrict__ agg) {
  const int tid = blockIdx.x * blockDim.x + threadIdx.x;
  if (tid < DD * DD) {
    const int k = tid >> 7;    // input dim 0..127
    const int n = tid & 127;   // output dim 0..127
    WpT[n * DD + k] = (_Float16)Wp[k * DD + n];
    WqT[n * DD + k] = (n < HD) ? (_Float16)Wq[k * HD + n] : (_Float16)0.f;
    WkT[n * DD + k] = (n < HD) ? (_Float16)Wk[k * HD + n] : (_Float16)0.f;
    WvT[n * DD + k] = (n < HD) ? (_Float16)Wv[k * HD + n] : (_Float16)0.f;
  }
  if (tid < DD) {
    bpP[tid] = bp[tid];
    bqP[tid] = (tid < HD) ? bq[tid] : 0.f;
    bkP[tid] = (tid < HD) ? bk[tid] : 0.f;
    bvP[tid] = (tid < HD) ? bv[tid] : 0.f;
    agg[tid] = 0.f;
  }
}

// ---------------------------------------------------------------------------
// 1) Embedding gather: X[t,n,:] = emb[ids[t,n],:]   (float4 per thread)
// ---------------------------------------------------------------------------
__global__ void gather_kernel(const int* __restrict__ ids,
                              const float* __restrict__ emb,
                              float* __restrict__ X) {
  const size_t tid = (size_t)blockIdx.x * blockDim.x + threadIdx.x;
  const size_t total = (size_t)TT * NN * (DD / 4);
  if (tid >= total) return;
  const size_t node = tid >> 5;          // which (t,n)
  const int c4 = (int)(tid & 31) << 2;   // 0,4,...,124
  const int id = ids[node];
  *(float4*)(X + node * DD + c4) = *(const float4*)(emb + (size_t)id * DD + c4);
}

// ---------------------------------------------------------------------------
// 2) Edge scatter: Nhat[t,dst,:] += X[t,src,:] * w   (Nhat pre-copied from X)
// ---------------------------------------------------------------------------
__global__ void edge_scatter_kernel(const int* __restrict__ src,
                                    const int* __restrict__ dst,
                                    const float* __restrict__ w,
                                    const float* __restrict__ X,
                                    float* __restrict__ Nhat) {
  const int b = blockIdx.x;      // t*E + e
  const int t = b >> 13;         // E = 8192
  const int s = src[b];
  const int d = dst[b];
  const float we = w[b];
  const float* xs = X + ((size_t)t * NN + s) * DD;
  float* nd       = Nhat + ((size_t)t * NN + d) * DD;
  atomicAdd(&nd[threadIdx.x], xs[threadIdx.x] * we);
}

// ---------------------------------------------------------------------------
// 3) Batched GEMM: Out[T*N,128] = A[T*N,128] @ W[128,128] + bias  (WMMA f16)
//    One block = 128x128 output tile; 8 waves; wave = one 16-row M strip.
//    B (f16 weights, 32 KB raw copy) staged via the Tensor Data Mover.
// ---------------------------------------------------------------------------
__global__ void __launch_bounds__(256) propagate_gemm_kernel(
    const float* __restrict__ Ain, const _Float16* __restrict__ Wt,
    const float* __restrict__ bias, float* __restrict__ Out) {
  __shared__ _Float16 Alds[128][128];   // 32 KB (f32 -> f16 staged)
  __shared__ _Float16 Blds[128][128];   // 32 KB, N-major: Blds[n][k]
  const int row0 = blockIdx.x * 128;

#if USE_TDM_ASM
  // Issue the B-matrix DMA first so it overlaps A staging. Wave 0 only.
  // TENSOR_LOAD_TO_LDS, 2-SGPR-group form (VADDR2/3 = NULL -> tensors <= 2D).
  if (threadIdx.x < 32) {
    // Generic pointers into LDS carry the LDS byte offset in the low dword
    // (ISA 10.2).
    const unsigned lds_b = (unsigned)(size_t)(void*)&Blds[0][0];
    const unsigned long long ga = (unsigned long long)(size_t)Wt;
    // D# group 0 (ISA 8.3): count=1 (valid), lds_addr, 57-bit global addr, type=2
    const u32x4 g0 = {1u, lds_b, (unsigned)ga,
                      (unsigned)((ga >> 32) & 0x1FFFFFFull) | (2u << 30)};
    // D# group 1 (ISA 8.4): data_size=2B; 1-D tile of 16384 halves (= 32 KB)
    const u32x8 g1 = {1u << 16,       // wg_mask=0 | data_size(2B)<<16
                      16384u << 16,   // tensor_dim0[15:0] in bits 63:48
                      1u << 16,       // tensor_dim0 hi=0 | tensor_dim1=1
                      16384u << 16,   // tensor_dim1 hi=0 | tile_dim0=16384
                      1u,             // tile_dim1=1 | tile_dim2=0
                      16384u, 0u, 0u};// tensor_dim0_stride=16384
    asm volatile("tensor_load_to_lds %0, %1" : : "s"(g0), "s"(g1) : "memory");
  }
#endif

  for (int i = threadIdx.x; i < 128 * 32; i += 256) {
    const int r = i >> 5;
    const int c4 = (i & 31) << 2;
    const float4 f = *(const float4*)(Ain + ((size_t)(row0 + r)) * DD + c4);
    store4h(&Alds[r][c4], f);
  }
#if USE_TDM_ASM
  if (threadIdx.x < 32) {
    asm volatile("s_wait_tensorcnt 0x0" : : : "memory");
  }
#else
  for (int i = threadIdx.x; i < 128 * 16; i += 256) {
    const int n = i >> 4;
    const int k8 = (i & 15) << 3;
    *(v8h*)&Blds[n][k8] = *(const v8h*)(Wt + (size_t)n * DD + k8);
  }
#endif
  __syncthreads();

  const int wave = threadIdx.x >> 5;
  const int lane = threadIdx.x & 31;
  const int m0 = wave * 16;
  v8f acc[8];
#pragma unroll
  for (int j = 0; j < 8; ++j) acc[j] = (v8f){};

#pragma unroll
  for (int kk = 0; kk < 4; ++kk) {
    const v16h a = load_a_frag(&Alds[0][0], lane, m0, kk * 32, 128);
#pragma unroll
    for (int j = 0; j < 8; ++j) {
      const v16h b = load_bt_frag(&Blds[0][0], lane, j * 16, kk * 32, 128);
      acc[j] = wmma_f16(a, b, acc[j]);
    }
  }

  const int nlo = lane & 15;
  const int mq = lane >> 4;  // C layout: M = r + 8*(lane>=16)
#pragma unroll
  for (int j = 0; j < 8; ++j) {
    const int n = j * 16 + nlo;
    const float bb = bias[n];
#pragma unroll
    for (int r = 0; r < 8; ++r) {
      const int m = m0 + r + 8 * mq;
      Out[((size_t)(row0 + m)) * DD + n] = acc[j][r] + bb;
    }
  }
}

// ---------------------------------------------------------------------------
// 4) Per-node temporal attention. One block (8 waves) per node.
//    QKV via WMMA, per-head scores via WMMA (Dh=10 zero-padded to K=32),
//    softmax, attn column-sums, agg[hd] += colsum @ V  (mean trick).
// ---------------------------------------------------------------------------
__global__ void __launch_bounds__(256) attention_kernel(
    const float* __restrict__ Xseq,                       // [T, N, 128]
    const _Float16* __restrict__ WqT, const float* __restrict__ bq,
    const _Float16* __restrict__ WkT, const float* __restrict__ bk,
    const _Float16* __restrict__ WvT, const float* __restrict__ bv,
    float* __restrict__ agg) {
  __shared__ _Float16 seq[64][128];       // 16 KB; reused as scores S[64][64] f32
  __shared__ _Float16 Qs[64 * HD];        // 15 KB each
  __shared__ _Float16 Ks[64 * HD];
  __shared__ _Float16 Vs[64 * HD];
  __shared__ float csum[64];
  __shared__ float aggl[128];
  float* S = (float*)&seq[0][0];

  const int node = blockIdx.x;
  const int tid = threadIdx.x;
  const int wave = tid >> 5;
  const int lane = tid & 31;

  for (int i = tid; i < 64 * 32; i += 256) {
    const int t = i >> 5;
    const int c4 = (i & 31) << 2;
    const float4 f = *(const float4*)(Xseq + ((size_t)t * NN + node) * DD + c4);
    store4h(&seq[t][c4], f);
  }
  if (tid < 128) aggl[tid] = 0.f;
  __syncthreads();

  // --- QKV: [64,128] @ [128,128]; 32 tiles per GEMM, 4 tiles per wave ---
#pragma unroll
  for (int which = 0; which < 3; ++which) {
    const _Float16* Wt = (which == 0) ? WqT : (which == 1) ? WkT : WvT;
    const float* bias  = (which == 0) ? bq  : (which == 1) ? bk  : bv;
    _Float16* Dst      = (which == 0) ? Qs  : (which == 1) ? Ks  : Vs;
#pragma unroll
    for (int i = 0; i < 4; ++i) {
      const int tileid = wave * 4 + i;
      const int m0 = (tileid >> 3) * 16;
      const int n0 = (tileid & 7) * 16;
      v8f acc = (v8f){};
#pragma unroll
      for (int kk = 0; kk < 4; ++kk) {
        const v16h a = load_a_frag(&seq[0][0], lane, m0, kk * 32, 128);
        const v16h b = load_bt_frag(Wt, lane, n0, kk * 32, 128);
        acc = wmma_f16(a, b, acc);
      }
      const int n = n0 + (lane & 15);
      if (n < HD) {
        const float bb = bias[n];
#pragma unroll
        for (int r = 0; r < 8; ++r) {
          const int m = m0 + r + 8 * (lane >> 4);
          Dst[m * HD + n] = (_Float16)(acc[r] + bb);
        }
      }
    }
  }
  __syncthreads();  // Q,K,V ready; seq no longer needed -> reuse as S

  const float scale = 0.31622776601683794f;  // 1/sqrt(10)
  for (int h = 0; h < HH; ++h) {
    __syncthreads();  // protect S/csum reuse across heads
    // scores S[t][s] = (Q_h @ K_h^T) * scale ; 16 tiles, 2 per wave, K=32 (10 valid)
#pragma unroll
    for (int i = 0; i < 2; ++i) {
      const int tileid = wave * 2 + i;
      const int m0 = (tileid >> 2) * 16;
      const int n0 = (tileid & 3) * 16;
      const int kh = lane >> 4;
      v16h a, b;
      const int m = m0 + (lane & 15);
#pragma unroll
      for (int j = 0; j < 16; ++j) {
        const int k = kh * 16 + j;
        a[j] = (k < DH) ? Qs[m * HD + h * DH + k] : (_Float16)0.f;
      }
      const int scol = n0 + (lane & 15);
#pragma unroll
      for (int j = 0; j < 16; ++j) {
        const int k = kh * 16 + j;
        b[j] = (k < DH) ? Ks[scol * HD + h * DH + k] : (_Float16)0.f;
      }
      v8f acc = (v8f){};
      acc = wmma_f16(a, b, acc);
#pragma unroll
      for (int r = 0; r < 8; ++r) {
        const int t = m0 + r + 8 * kh;
        S[t * 64 + scol] = acc[r] * scale;
      }
    }
    __syncthreads();
    // softmax over s (rows handled by threads 0..63)
    if (tid < 64) {
      float mx = -1e30f;
      for (int s2 = 0; s2 < 64; ++s2) mx = fmaxf(mx, S[tid * 64 + s2]);
      float sum = 0.f;
      for (int s2 = 0; s2 < 64; ++s2) {
        const float e = __expf(S[tid * 64 + s2] - mx);
        S[tid * 64 + s2] = e;
        sum += e;
      }
      const float inv = 1.f / sum;
      for (int s2 = 0; s2 < 64; ++s2) S[tid * 64 + s2] *= inv;
    }
    __syncthreads();
    if (tid < 64) {  // column sums of attn
      float c = 0.f;
      for (int t2 = 0; t2 < 64; ++t2) c += S[t2 * 64 + tid];
      csum[tid] = c;
    }
    __syncthreads();
    if (tid < DH) {  // aggl[h*10+d] += colsum @ V_h
      float a2 = 0.f;
      for (int s2 = 0; s2 < 64; ++s2)
        a2 += csum[s2] * (float)Vs[s2 * HD + h * DH + tid];
      aggl[h * DH + tid] += a2;
    }
  }
  __syncthreads();
  if (tid < HD) atomicAdd(&agg[tid], aggl[tid]);
}

// ---------------------------------------------------------------------------
// 5) Finalize: out[j] = (agg/(N*T)) @ Wo + bo
// ---------------------------------------------------------------------------
__global__ void finalize_kernel(const float* __restrict__ agg,
                                const float* __restrict__ Wo,
                                const float* __restrict__ bo,
                                float* __restrict__ out) {
  const int j = threadIdx.x;
  if (j < 2) {
    float s = 0.f;
    for (int i = 0; i < HD; ++i) s += agg[i] * Wo[i * 2 + j];
    out[j] = s * (1.f / ((float)NN * (float)TT)) + bo[j];
  }
}

// ---------------------------------------------------------------------------
extern "C" void kernel_launch(void* const* d_in, const int* in_sizes, int n_in,
                              void* d_out, int out_size, void* d_ws, size_t ws_size,
                              hipStream_t stream) {
  const int*   node_ids = (const int*)d_in[0];
  const int*   src      = (const int*)d_in[1];
  const int*   dst      = (const int*)d_in[2];
  const float* ew       = (const float*)d_in[3];
  const float* emb      = (const float*)d_in[4];
  const float* Wp       = (const float*)d_in[5];
  const float* bp       = (const float*)d_in[6];
  const float* Wq       = (const float*)d_in[7];
  const float* bq       = (const float*)d_in[8];
  const float* Wk       = (const float*)d_in[9];
  const float* bk       = (const float*)d_in[10];
  const float* Wv       = (const float*)d_in[11];
  const float* bv       = (const float*)d_in[12];
  const float* Wo       = (const float*)d_in[13];
  const float* bo       = (const float*)d_in[14];
  float* out = (float*)d_out;

  // workspace layout
  char* ws = (char*)d_ws;
  const size_t stateBytes = (size_t)TT * NN * DD * sizeof(float);  // 64 MB
  size_t off = 0;
  float* bufA = (float*)(ws + off); off += stateBytes;   // x / node_emb
  float* bufB = (float*)(ws + off); off += stateBytes;   // nhat scratch
  _Float16* WpT = (_Float16*)(ws + off); off += DD * DD * sizeof(_Float16);
  _Float16* WqT = (_Float16*)(ws + off); off += DD * DD * sizeof(_Float16);
  _Float16* WkT = (_Float16*)(ws + off); off += DD * DD * sizeof(_Float16);
  _Float16* WvT = (_Float16*)(ws + off); off += DD * DD * sizeof(_Float16);
  float* bpP = (float*)(ws + off); off += DD * sizeof(float);
  float* bqP = (float*)(ws + off); off += DD * sizeof(float);
  float* bkP = (float*)(ws + off); off += DD * sizeof(float);
  float* bvP = (float*)(ws + off); off += DD * sizeof(float);
  float* agg = (float*)(ws + off); off += DD * sizeof(float);

  // 0) weight prep (+ zero agg)
  prep_kernel<<<64, 256, 0, stream>>>(Wp, bp, Wq, bq, Wk, bk, Wv, bv,
                                      WpT, WqT, WkT, WvT, bpP, bqP, bkP, bvP, agg);
  // 1) embedding gather
  gather_kernel<<<(TT * NN * (DD / 4)) / 256, 256, 0, stream>>>(node_ids, emb, bufA);
  // 2-3) three propagate rounds: nhat = x + scatter(msgs); x = nhat @ Wp + bp
  for (int r = 0; r < 3; ++r) {
    hipMemcpyAsync(bufB, bufA, stateBytes, hipMemcpyDeviceToDevice, stream);
    edge_scatter_kernel<<<TT * EE, 128, 0, stream>>>(src, dst, ew, bufA, bufB);
    propagate_gemm_kernel<<<(TT * NN) / 128, 256, 0, stream>>>(bufB, WpT, bpP, bufA);
  }
  // 4) per-node attention + agg accumulation
  attention_kernel<<<NN, 256, 0, stream>>>(bufA, WqT, bqP, WkT, bkP, WvT, bvP, agg);
  // 5) final 120 -> 2 projection
  finalize_kernel<<<1, 64, 0, stream>>>(agg, Wo, bo, out);
}
